// Reactnet_52381421142581
// MI455X (gfx1250) — 8-bit matrix operand order repaired and checked by reading, compiled, not run on hardware
//
#include <hip/hip_runtime.h>
#include <hip/hip_bf16.h>
#include <cstdint>
#include <cstddef>


typedef __attribute__((ext_vector_type(8))) int v8i;

#define EPSBN 1e-5f

__global__ void __launch_bounds__(256)
firstconv_kernel(const float* __restrict__ x,
                 const float* __restrict__ w,
                 const float* __restrict__ mul, const float* __restrict__ add,
                 float* __restrict__ y)
{
    int t = blockIdx.x * blockDim.x + threadIdx.x;
    if (t >= 256 * 32 * 32 * 32) return;
    int co = t & 31, wp = (t >> 5) & 31, hp = (t >> 10) & 31, n = t >> 15;
    float acc = 0.f;
    #pragma unroll
    for (int ci = 0; ci < 3; ++ci)
        #pragma unroll
        for (int ky = 0; ky < 3; ++ky) {
            int hi = hp + ky - 1;
            if (hi < 0 || hi >= 32) continue;
            #pragma unroll
            for (int kx = 0; kx < 3; ++kx) {
                int wi = wp + kx - 1;
                if (wi < 0 || wi >= 32) continue;
                acc += x[((n * 3 + ci) * 32 + hi) * 32 + wi] *
                       w[((co * 3 + ci) * 3 + ky) * 3 + kx];
            }
        }
    y[t] = acc * mul[co] + add[co];
}

__global__ void __launch_bounds__(256)
im2col_sign_kernel(const float* __restrict__ x,
                   const float* __restrict__ move,
                   int8_t* __restrict__ A,
                   int H, int W, int C,
                   int Ho, int Wo, int stride, int ksz,
                   int K, int KP)
{
    long idx = (long)blockIdx.x * blockDim.x + threadIdx.x;
    int  kq4 = KP >> 2;
    long total = (long)256 * Ho * Wo * kq4;
    if (idx >= total) return;
    int  kq = (int)(idx % kq4);
    long m  = idx / kq4;
    int  k0 = kq << 2;
    int packed = 0;
    if (k0 < K) {
        int wo = (int)(m % Wo);
        long tmp = m / Wo;
        int ho = (int)(tmp % Ho);
        int n  = (int)(tmp / Ho);
        #pragma unroll
        for (int j = 0; j < 4; ++j) {
            int k = k0 + j;
            int c, hi, wi;
            if (ksz == 3) {
                int tap = k / C; c = k - tap * C;
                int ky = tap / 3, kx = tap - ky * 3;
                hi = ho * stride + ky - 1; wi = wo * stride + kx - 1;
            } else { c = k; hi = ho; wi = wo; }
            int s = 0;
            if (hi >= 0 && hi < H && wi >= 0 && wi < W) {
                float v = x[(((long)n * H + hi) * W + wi) * C + c] + move[c];
                s = (v > 0.f) ? 1 : ((v < 0.f) ? -1 : 0);
            }
            packed |= (s & 0xFF) << (8 * j);
        }
    }
    *(int*)(A + m * KP + k0) = packed;
}

__global__ void wscale_kernel(const float* __restrict__ w, float* __restrict__ scale,
                              int Cout, int Kw)
{
    int c = blockIdx.x * blockDim.x + threadIdx.x;
    if (c >= Cout) return;
    const float* p = w + (long)c * Kw;
    float s = 0.f;
    for (int i = 0; i < Kw; ++i) s += fabsf(p[i]);
    scale[c] = s / (float)Kw;
}

__global__ void bnfold_kernel(const float* __restrict__ scale,
                              const float* __restrict__ g, const float* __restrict__ b,
                              const float* __restrict__ m, const float* __restrict__ v,
                              float* __restrict__ mul, float* __restrict__ add, int C)
{
    int c = blockIdx.x * blockDim.x + threadIdx.x;
    if (c >= C) return;
    float inv = g[c] * rsqrtf(v[c] + EPSBN);
    mul[c] = (scale ? scale[c] : 1.f) * inv;
    add[c] = b[c] - m[c] * inv;
}

__global__ void __launch_bounds__(256)
wpack_kernel(const float* __restrict__ w,
             int* __restrict__ Bp,
             int Cout, int Cin, int ksz, int K, int KP)
{
    int KB = KP >> 6;
    long t = (long)blockIdx.x * blockDim.x + threadIdx.x;
    long total = (long)(Cout >> 4) * KB * 32 * 8;
    if (t >= total) return;
    int r    = (int)(t & 7);
    int lane = (int)((t >> 3) & 31);
    long hi  = t >> 8;
    int kb    = (int)(hi % KB);
    int ntile = (int)(hi / KB);
    int khalf = lane >> 4;
    int n     = ntile * 16 + (lane & 15);
    int kloc  = ((r >= 4) ? 32 : 0) + (khalf << 4) + ((r & 3) << 2);
    int Kw    = Cin * ksz * ksz;
    int packed = 0;
    #pragma unroll
    for (int j = 0; j < 4; ++j) {
        int k = (kb << 6) + kloc + j;
        int s = 0;
        if (k < K) {
            int widx;
            if (ksz == 3) { int tap = k / Cin; int c = k - tap * Cin; widx = c * 9 + tap; }
            else          { widx = k; }
            float v = w[(long)n * Kw + widx];
            s = (v > 0.f) ? 1 : ((v < 0.f) ? -1 : 0);
        }
        packed |= (s & 0xFF) << (8 * j);
    }
    Bp[t] = packed;
}

template<int NT>
__global__ void __launch_bounds__(256)
bgemm_iu8_kernel(const int8_t* __restrict__ A,
                 const int*    __restrict__ Bp,
                 int*          __restrict__ Cg,
                 int M, int KP, int Nout)
{
    const int lane  = threadIdx.x & 31;
    const int mtile = blockIdx.x * 8 + (threadIdx.x >> 5);
    const int ng    = blockIdx.y;
    if (mtile * 16 >= M) return;
    const int khalf = lane >> 4;
    const int mrow  = mtile * 16 + (lane & 15);
    const int KB    = KP >> 6;
    const long arow = (long)mrow * KP;

    const int4* bptr[NT];
    #pragma unroll
    for (int t = 0; t < NT; ++t)
        bptr[t] = (const int4*)(Bp + (((long)(ng * NT + t) * KB) * 32 + lane) * 8);

    v8i acc[NT];
    #pragma unroll
    for (int t = 0; t < NT; ++t) { v8i z = {}; acc[t] = z; }

    auto loadA = [&](int kb, v8i& a) {
        const int kb64 = kb << 6;
        #pragma unroll
        for (int r = 0; r < 8; ++r) {
            int g = r & 3;
            int kbase = ((r >> 2) << 5) + (khalf << 4) + (g << 2);
            a[r] = *(const int*)(A + arow + kb64 + kbase);
        }
    };
    auto loadB = [&](int kb, int t, v8i& b) {
        const int4* p = bptr[t] + (long)kb * 64;
        int4 b0 = p[0], b1 = p[1];
        b[0] = b0.x; b[1] = b0.y; b[2] = b0.z; b[3] = b0.w;
        b[4] = b1.x; b[5] = b1.y; b[6] = b1.z; b[7] = b1.w;
    };

    v8i aCur, bCur[NT];
    loadA(0, aCur);
    #pragma unroll
    for (int t = 0; t < NT; ++t) loadB(0, t, bCur[t]);

    #pragma unroll 2
    for (int kb = 0; kb < KB - 1; ++kb) {
        v8i aNxt, bNxt[NT];
        __builtin_prefetch(A + arow + ((kb + 2) << 6), 0, 1);
        loadA(kb + 1, aNxt);
        #pragma unroll
        for (int t = 0; t < NT; ++t) loadB(kb + 1, t, bNxt[t]);
        #pragma unroll
        for (int t = 0; t < NT; ++t)
            acc[t] = __builtin_amdgcn_wmma_i32_16x16x64_iu8(
                true, aCur, true, bCur[t], acc[t], false, false);
        aCur = aNxt;
        #pragma unroll
        for (int t = 0; t < NT; ++t) bCur[t] = bNxt[t];
    }

    #pragma unroll
    for (int t = 0; t < NT; ++t)
        acc[t] = __builtin_amdgcn_wmma_i32_16x16x64_iu8(
            true, aCur, true, bCur[t], acc[t], false, false);

    #pragma unroll
    for (int t = 0; t < NT; ++t) {
        const int ncol = (ng * NT + t) * 16 + (lane & 15);
        #pragma unroll
        for (int r = 0; r < 8; ++r) {
            int row = mtile * 16 + r + (khalf << 3);
            Cg[(long)row * Nout + ncol] = acc[t][r];
        }
    }
}

__global__ void avgpool_kernel(const float* __restrict__ x, float* __restrict__ y,
                               int H, int W, int C)
{
    int Ho = H >> 1, Wo = W >> 1;
    long t = (long)blockIdx.x * blockDim.x + threadIdx.x;
    long total = (long)256 * Ho * Wo * C;
    if (t >= total) return;
    int c = (int)(t % C);
    long r = t / C;
    int wo = (int)(r % Wo); r /= Wo;
    int ho = (int)(r % Ho);
    int n  = (int)(r / Ho);
    long base = (((long)n * H + 2 * ho) * W + 2 * wo) * C + c;
    long rowC = (long)W * C;
    y[t] = 0.25f * (x[base] + x[base + C] + x[base + rowC] + x[base + rowC + C]);
}

__global__ void __launch_bounds__(256)
epilogue_kernel(const int* __restrict__ g, const float* __restrict__ mul,
                const float* __restrict__ add, const float* __restrict__ resid,
                const float* __restrict__ mvA, const float* __restrict__ a,
                const float* __restrict__ mvB, float* __restrict__ out,
                long total, int C)
{
    long t = (long)blockIdx.x * blockDim.x + threadIdx.x;
    if (t >= total) return;
    int c = (int)(t % C);
    float y = (float)g[t] * mul[c] + add[c] + resid[t];
    float u = y + mvA[c];
    out[t] = ((u >= 0.f) ? u : a[c] * u) + mvB[c];
}

__global__ void __launch_bounds__(256)
epilogue_cat_kernel(const int* __restrict__ ga, const int* __restrict__ gb,
                    const float* __restrict__ mulA, const float* __restrict__ addA,
                    const float* __restrict__ mulB, const float* __restrict__ addB,
                    const float* __restrict__ out1,
                    const float* __restrict__ mv22, const float* __restrict__ a2,
                    const float* __restrict__ mv23, float* __restrict__ out,
                    long M, int C)
{
    long t = (long)blockIdx.x * blockDim.x + threadIdx.x;
    long total = M * 2 * C;
    if (t >= total) return;
    int c2 = (int)(t % (2 * C));
    long m = t / (2 * C);
    float v;
    if (c2 < C)
        v = (float)ga[m * C + c2] * mulA[c2] + addA[c2] + out1[m * C + c2];
    else {
        int c = c2 - C;
        v = (float)gb[m * C + c] * mulB[c] + addB[c] + out1[m * C + c];
    }
    float u = v + mv22[c2];
    out[t] = ((u >= 0.f) ? u : a2[c2] * u) + mv23[c2];
}

__global__ void gap_kernel(const float* __restrict__ x, float* __restrict__ p,
                           int HW, int C)
{
    int t = blockIdx.x * blockDim.x + threadIdx.x;
    if (t >= 256 * C) return;
    int c = t % C, n = t / C;
    float s = 0.f;
    for (int i = 0; i < HW; ++i) s += x[((long)n * HW + i) * C + c];
    p[t] = s / (float)HW;
}

__global__ void fc_kernel(const float* __restrict__ p, const float* __restrict__ w,
                          const float* __restrict__ b, float* __restrict__ out)
{
    int t = blockIdx.x * blockDim.x + threadIdx.x;
    if (t >= 2560) return;
    int j = t % 10, n = t / 10;
    float s = b[j];
    const float* pr = p + (long)n * 1024;
    const float* wr = w + (long)j * 1024;
    for (int i = 0; i < 1024; ++i) s += pr[i] * wr[i];
    out[t] = s;
}

extern "C" void kernel_launch(void* const* d_in, const int* in_sizes, int n_in,
                              void* d_out, int out_size, void* d_ws, size_t ws_size,
                              hipStream_t stream)
{
    (void)in_sizes; (void)n_in; (void)out_size; (void)ws_size;

    static const int STAGE[14] = {32,64,128,128,256,256,512,512,512,512,512,512,1024,1024};

    int cur = 0;
    auto nxt = [&]() { return (const float*)d_in[cur++]; };
    const float* x_in    = nxt();
    const float* first_w = nxt();
    const float* fbg = nxt(); const float* fbb = nxt();
    const float* fbm = nxt(); const float* fbv = nxt();

    struct BP {
        const float *mv11,*w3,*b1g,*b1b,*b1m,*b1v,*mv12,*a1,*mv13,*mv21,*mv22,*a2,*mv23;
        const float *pw,*b2g,*b2b,*b2m,*b2v;
        const float *pw1,*b21g,*b21b,*b21m,*b21v;
        const float *pw2,*b22g,*b22b,*b22m,*b22v;
    };
    BP bp[13]; int cin_[13], cout_[13], str_[13];
    for (int i = 0; i < 13; ++i) {
        cin_[i]  = STAGE[i];
        cout_[i] = STAGE[i + 1];
        str_[i]  = (STAGE[i] != STAGE[i + 1] && STAGE[i + 1] != 64) ? 2 : 1;
        BP& p = bp[i];
        p.mv11 = nxt(); p.w3 = nxt();
        p.b1g = nxt(); p.b1b = nxt(); p.b1m = nxt(); p.b1v = nxt();
        p.mv12 = nxt(); p.a1 = nxt(); p.mv13 = nxt();
        p.mv21 = nxt(); p.mv22 = nxt(); p.a2 = nxt(); p.mv23 = nxt();
        if (cin_[i] == cout_[i]) {
            p.pw = nxt();
            p.b2g = nxt(); p.b2b = nxt(); p.b2m = nxt(); p.b2v = nxt();
        } else {
            p.pw1 = nxt();
            p.b21g = nxt(); p.b21b = nxt(); p.b21m = nxt(); p.b21v = nxt();
            p.pw2 = nxt();
            p.b22g = nxt(); p.b22b = nxt(); p.b22m = nxt(); p.b22v = nxt();
        }
    }
    const float* fcw = nxt();
    const float* fcb = nxt();

    const size_t MB = 1u << 20;
    char* ws = (char*)d_ws;
    float*  actA  = (float*)(ws + 0 * MB);
    float*  actB  = (float*)(ws + 64 * MB);
    float*  out1  = (float*)(ws + 128 * MB);
    float*  xpool = (float*)(ws + 192 * MB);
    int8_t* Abuf  = (int8_t*)(ws + 208 * MB);
    int*    gemA  = (int*)(ws + 288 * MB);
    int*    gemB  = (int*)(ws + 352 * MB);
    int*    wpA   = (int*)(ws + 384 * MB);
    int*    wpB   = (int*)(ws + 400 * MB);
    float*  fbuf  = (float*)(ws + 404 * MB);
    float* scaleA = fbuf;          float* scaleB = fbuf + 2048;
    float* mulA   = fbuf + 4096;   float* addA   = fbuf + 6144;
    float* mulB   = fbuf + 8192;   float* addB   = fbuf + 10240;

    auto g1 = [](long n, int b) { return dim3((unsigned)((n + b - 1) / b)); };

    auto launch_bgemm = [&](const int8_t* Ab, const int* Wp, int* Cg,
                            long M, int KP, int Nout) {
        unsigned gx = (unsigned)((M / 16 + 7) / 8);
        if ((Nout & 63) == 0) {
            dim3 gg(gx, (unsigned)(Nout / 64));
            bgemm_iu8_kernel<4><<<gg, 256, 0, stream>>>(Ab, Wp, Cg, (int)M, KP, Nout);
        } else if ((Nout & 31) == 0) {
            dim3 gg(gx, (unsigned)(Nout / 32));
            bgemm_iu8_kernel<2><<<gg, 256, 0, stream>>>(Ab, Wp, Cg, (int)M, KP, Nout);
        } else {
            dim3 gg(gx, (unsigned)(Nout / 16));
            bgemm_iu8_kernel<1><<<gg, 256, 0, stream>>>(Ab, Wp, Cg, (int)M, KP, Nout);
        }
    };

    bnfold_kernel<<<g1(32, 64), 64, 0, stream>>>(nullptr, fbg, fbb, fbm, fbv,
                                                 mulA, addA, 32);
    firstconv_kernel<<<g1(8388608, 256), 256, 0, stream>>>(x_in, first_w, mulA, addA, actA);

    int H = 32, W = 32;
    for (int i = 0; i < 13; ++i) {
        const BP& p = bp[i];
        const int cin = cin_[i], cout = cout_[i], s = str_[i];
        const int Ho = H / s, Wo = W / s;
        const long M = 256L * Ho * Wo;

        const int K1 = 9 * cin, KP1 = (K1 + 63) & ~63;
        im2col_sign_kernel<<<g1(M * (KP1 >> 2), 256), 256, 0, stream>>>(
            actA, p.mv11, Abuf, H, W, cin, Ho, Wo, s, 3, K1, KP1);
        wscale_kernel<<<g1(cin, 64), 64, 0, stream>>>(p.w3, scaleA, cin, cin * 9);
        wpack_kernel<<<g1((long)(cin >> 4) * (KP1 >> 6) * 256, 256), 256, 0, stream>>>(
            p.w3, wpA, cin, cin, 3, K1, KP1);
        bnfold_kernel<<<g1(cin, 64), 64, 0, stream>>>(scaleA, p.b1g, p.b1b, p.b1m, p.b1v,
                                                      mulA, addA, cin);
        launch_bgemm(Abuf, wpA, gemA, M, KP1, cin);

        const float* resid = actA;
        if (s == 2) {
            avgpool_kernel<<<g1(M * cin, 256), 256, 0, stream>>>(actA, xpool, H, W, cin);
            resid = xpool;
        }
        epilogue_kernel<<<g1(M * cin, 256), 256, 0, stream>>>(
            gemA, mulA, addA, resid, p.mv12, p.a1, p.mv13, out1, M * cin, cin);

        const int KP2 = (cin + 63) & ~63;
        im2col_sign_kernel<<<g1(M * (KP2 >> 2), 256), 256, 0, stream>>>(
            out1, p.mv21, Abuf, Ho, Wo, cin, Ho, Wo, 1, 1, cin, KP2);
        if (cin == cout) {
            wscale_kernel<<<g1(cout, 64), 64, 0, stream>>>(p.pw, scaleA, cout, cin);
            wpack_kernel<<<g1((long)(cout >> 4) * (KP2 >> 6) * 256, 256), 256, 0, stream>>>(
                p.pw, wpA, cout, cin, 1, cin, KP2);
            bnfold_kernel<<<g1(cout, 64), 64, 0, stream>>>(scaleA, p.b2g, p.b2b, p.b2m, p.b2v,
                                                           mulA, addA, cout);
            launch_bgemm(Abuf, wpA, gemA, M, KP2, cout);
            epilogue_kernel<<<g1(M * cout, 256), 256, 0, stream>>>(
                gemA, mulA, addA, out1, p.mv22, p.a2, p.mv23, actB, M * cout, cout);
        } else {
            wscale_kernel<<<g1(cin, 64), 64, 0, stream>>>(p.pw1, scaleA, cin, cin);
            wscale_kernel<<<g1(cin, 64), 64, 0, stream>>>(p.pw2, scaleB, cin, cin);
            wpack_kernel<<<g1((long)(cin >> 4) * (KP2 >> 6) * 256, 256), 256, 0, stream>>>(
                p.pw1, wpA, cin, cin, 1, cin, KP2);
            wpack_kernel<<<g1((long)(cin >> 4) * (KP2 >> 6) * 256, 256), 256, 0, stream>>>(
                p.pw2, wpB, cin, cin, 1, cin, KP2);
            bnfold_kernel<<<g1(cin, 64), 64, 0, stream>>>(scaleA, p.b21g, p.b21b, p.b21m, p.b21v,
                                                          mulA, addA, cin);
            bnfold_kernel<<<g1(cin, 64), 64, 0, stream>>>(scaleB, p.b22g, p.b22b, p.b22m, p.b22v,
                                                          mulB, addB, cin);
            launch_bgemm(Abuf, wpA, gemA, M, KP2, cin);
            launch_bgemm(Abuf, wpB, gemB, M, KP2, cin);
            epilogue_cat_kernel<<<g1(M * 2 * cin, 256), 256, 0, stream>>>(
                gemA, gemB, mulA, addA, mulB, addB, out1,
                p.mv22, p.a2, p.mv23, actB, M, cin);
        }
        { float* t = actA; actA = actB; actB = t; }
        H = Ho; W = Wo;
    }

    gap_kernel<<<g1(256 * 1024, 256), 256, 0, stream>>>(actA, xpool, H * W, 1024);
    fc_kernel<<<g1(2560, 256), 256, 0, stream>>>(xpool, fcw, fcb, (float*)d_out);
}
